// EncoderConcat_84104049590407
// MI455X (gfx1250) — matrix operand order repaired and checked by reading, not compiled, not run on hardware
//
#include <hip/hip_runtime.h>

// ---------------------------------------------------------------------------
// CDNA5 (gfx1250) fused GraphCast g2m encoder, round 2.
// 32 rows per block (2 M-tiles per wave) so every B fragment load feeds two
// v_wmma_f32_16x16x32_bf16 ops; float4-vectorized gather; bf16 weights
// pre-transposed to [N][K]; scatter-sum via global_atomic_add_f32.
// ---------------------------------------------------------------------------

typedef __bf16 bf16_t;
typedef __bf16 v16bf __attribute__((ext_vector_type(16)));
typedef __bf16 v8bf  __attribute__((ext_vector_type(8)));
typedef __bf16 v4bf  __attribute__((ext_vector_type(4)));
typedef float  v8f   __attribute__((ext_vector_type(8)));

#define D_     512
#define H_     512
#define E_     260640
#define NG_    65160
#define NM_    10242
#define LN_EPS 1e-5f

// ---------------- workspace layout (bytes) ----------------
#define OFF_EW1 0ull                                   // 512*1536 bf16
#define OFF_EW2 (OFF_EW1 + 512ull*1536*2)              // 512*512 bf16
#define OFF_SW1 (OFF_EW2 + 512ull*512*2)
#define OFF_SW2 (OFF_SW1 + 512ull*512*2)
#define OFF_DW1 (OFF_SW2 + 512ull*512*2)               // 512*1024 bf16
#define OFF_DW2 (OFF_DW1 + 512ull*1024*2)
#define OFF_AGG (OFF_DW2 + 512ull*512*2)               // NM*512 f32

// ---------------------------------------------------------------------------
// 32xK @ Kx512 GEMM sweep: wave computes 2 M-tiles x 8 N-tiles. Each B
// fragment (one contiguous 32B load from transposed weights) is reused by
// both M-tiles; A fragments come from LDS (two 16B loads each).
// ---------------------------------------------------------------------------
__device__ __forceinline__ void wmma_gemm32(const bf16_t* __restrict__ Ash, int lda,
                                            const bf16_t* __restrict__ Wt, int K,
                                            int wave, int lane, v8f acc[2][8]) {
  const int m  = lane & 15;   // A: row-in-tile / B: column
  const int hh = lane >> 4;   // half select
  for (int kk = 0; kk < K; kk += 32) {
    v16bf a[2];
#pragma unroll
    for (int mt = 0; mt < 2; ++mt) {
      const bf16_t* ap = Ash + (mt * 16 + m) * lda + kk + hh * 16;
      v8bf lo = *(const v8bf*)(ap);
      v8bf hi = *(const v8bf*)(ap + 8);
      a[mt] = __builtin_shufflevector(lo, hi, 0,1,2,3,4,5,6,7,8,9,10,11,12,13,14,15);
    }
#pragma unroll
    for (int t = 0; t < 8; ++t) {
      const int ntile = wave * 8 + t;
      const bf16_t* bp = Wt + (size_t)(ntile * 16 + m) * (size_t)K + kk + hh * 16;
      v16bf b = *(const v16bf*)bp;
      acc[0][t] = __builtin_amdgcn_wmma_f32_16x16x32_bf16(
          false, a[0], false, b, (short)0, acc[0][t], false, false);
      acc[1][t] = __builtin_amdgcn_wmma_f32_16x16x32_bf16(
          false, a[1], false, b, (short)0, acc[1][t], false, false);
    }
  }
}

// bias + SiLU, write 32x512 hidden tile to LDS as bf16 (C/D lane layout).
__device__ __forceinline__ void store_hidden_silu(bf16_t* Hsh, const v8f acc[2][8],
                                                  const float* __restrict__ b1,
                                                  int wave, int lane) {
  const int n = lane & 15, hh = lane >> 4;
#pragma unroll
  for (int t = 0; t < 8; ++t) {
    const int col = (wave * 8 + t) * 16 + n;
    const float bv = b1[col];
#pragma unroll
    for (int mt = 0; mt < 2; ++mt)
#pragma unroll
      for (int r = 0; r < 8; ++r) {
        float x = acc[mt][t][r] + bv;
        float s = x / (1.0f + __expf(-x));        // SiLU
        Hsh[(mt * 16 + hh * 8 + r) * H_ + col] = (bf16_t)s;
      }
  }
}

// bias, write 32x512 f32 tile to LDS.
__device__ __forceinline__ void store_y(float* Ysh, const v8f acc[2][8],
                                        const float* __restrict__ b2,
                                        int wave, int lane) {
  const int n = lane & 15, hh = lane >> 4;
#pragma unroll
  for (int t = 0; t < 8; ++t) {
    const int col = (wave * 8 + t) * 16 + n;
    const float bv = b2[col];
#pragma unroll
    for (int mt = 0; mt < 2; ++mt)
#pragma unroll
      for (int r = 0; r < 8; ++r)
        Ysh[(mt * 16 + hh * 8 + r) * D_ + col] = acc[mt][t][r] + bv;
  }
}

// Full 2-layer MLP core on a 32-row tile already staged in Ash.
template <int DIN>
__device__ __forceinline__ void mlp_core(const bf16_t* Ash, bf16_t* Hsh, float* Ysh,
                                         const bf16_t* W1t, const float* b1,
                                         const bf16_t* W2t, const float* b2, int tid) {
  const int lane = tid & 31, wave = tid >> 5;
  v8f acc[2][8];
#pragma unroll
  for (int mt = 0; mt < 2; ++mt)
#pragma unroll
    for (int t = 0; t < 8; ++t) acc[mt][t] = (v8f){0,0,0,0,0,0,0,0};
  wmma_gemm32(Ash, DIN, W1t, DIN, wave, lane, acc);
  __syncthreads();                       // Ash reads done before Hsh (may alias)
  store_hidden_silu(Hsh, acc, b1, wave, lane);
  __syncthreads();
#pragma unroll
  for (int mt = 0; mt < 2; ++mt)
#pragma unroll
    for (int t = 0; t < 8; ++t) acc[mt][t] = (v8f){0,0,0,0,0,0,0,0};
  wmma_gemm32(Hsh, H_, W2t, H_, wave, lane, acc);
  store_y(Ysh, acc, b2, wave, lane);     // Ysh never aliases Hsh
  __syncthreads();
}

// LayerNorm stats: 128 threads <-> 32 rows x 4 segments of 128 cols.
__device__ __forceinline__ float2 ln_stats32(const float* Ysh, float* red, int tid) {
  const int row = tid >> 2, seg = tid & 3;
  float s = 0.f, ss = 0.f;
  const float* yr = Ysh + row * D_ + seg * 128;
#pragma unroll 8
  for (int c = 0; c < 128; ++c) { float v = yr[c]; s += v; ss += v * v; }
  red[row * 4 + seg] = s;
  red[128 + row * 4 + seg] = ss;
  __syncthreads();
  float mu = 0.f, m2 = 0.f;
#pragma unroll
  for (int i = 0; i < 4; ++i) { mu += red[row * 4 + i]; m2 += red[128 + row * 4 + i]; }
  mu *= (1.0f / 512.0f);
  m2 *= (1.0f / 512.0f);
  float inv = rsqrtf(m2 - mu * mu + LN_EPS);
  return make_float2(mu, inv);
}

// ---------------------------------------------------------------------------
// Kernel 0: fp32 W[din][dout] -> bf16 Wt[dout][din]
// ---------------------------------------------------------------------------
__global__ void convert_wt_kernel(const float* __restrict__ W, bf16_t* __restrict__ Wt,
                                  int din, int dout) {
  int i = blockIdx.x * 256 + threadIdx.x;
  if (i < din * dout) {
    int n = i / din, k = i - n * din;
    Wt[i] = (bf16_t)W[(size_t)k * dout + n];
  }
}

__global__ void zero_kernel(float* __restrict__ p, int n) {
  int i = blockIdx.x * 256 + threadIdx.x;
  if (i < n) p[i] = 0.0f;
}

// ---------------------------------------------------------------------------
// Kernel 1: edge MLP (gather-concat -> MLP -> LN -> atomic scatter to agg)
// 32 edges per block.
// ---------------------------------------------------------------------------
__global__ __launch_bounds__(128) void edge_mlp_kernel(
    const float* __restrict__ g2m, const float* __restrict__ grid,
    const float* __restrict__ mesh, const int* __restrict__ src_idx,
    const int* __restrict__ dst_idx, const bf16_t* __restrict__ W1t,
    const float* __restrict__ b1, const bf16_t* __restrict__ W2t,
    const float* __restrict__ b2, const float* __restrict__ g,
    const float* __restrict__ beta, float* __restrict__ agg) {
  __shared__ alignas(32) unsigned char smem[98304];
  __shared__ float red[256];
  __shared__ int sidx[32], didx[32];
  bf16_t* Ash = (bf16_t*)smem;               // [32][1536] during GEMM1
  bf16_t* Hsh = (bf16_t*)smem;               // [32][512]  (aliases Ash head)
  float*  Ysh = (float*)(smem + 32768);      // [32][512]  (aliases Ash tail)

  const int tid = threadIdx.x;
  const size_t e0 = (size_t)blockIdx.x * 32;
  if (tid < 32) {
    sidx[tid] = src_idx[e0 + tid];
    didx[tid] = dst_idx[e0 + tid];
  }
  __syncthreads();

  // gather-concat [32][1536] -> bf16 LDS, float4-vectorized
  for (int j = tid; j < 32 * 384; j += 128) {     // 384 float4 per row
    int row = j / 384, c4 = j - row * 384;
    int col = c4 * 4;
    const float* src;
    if (col < 512)       src = g2m  + (e0 + row) * D_ + col;
    else if (col < 1024) src = grid + (size_t)sidx[row] * D_ + (col - 512);
    else                 src = mesh + (size_t)didx[row] * D_ + (col - 1024);
    float4 f = *(const float4*)src;
    v4bf o = { (bf16_t)f.x, (bf16_t)f.y, (bf16_t)f.z, (bf16_t)f.w };
    *(v4bf*)(Ash + row * 1536 + col) = o;
  }
  __syncthreads();

  mlp_core<1536>(Ash, Hsh, Ysh, W1t, b1, W2t, b2, tid);

  float2 st = ln_stats32(Ysh, red, tid);
  const int row = tid >> 2, seg = tid & 3;
  const size_t dbase = (size_t)didx[row] * D_;
#pragma unroll 4
  for (int c = 0; c < 128; ++c) {
    int col = seg * 128 + c;
    float v = (Ysh[row * D_ + col] - st.x) * st.y * g[col] + beta[col];
    unsafeAtomicAdd(&agg[dbase + col], v);   // global_atomic_add_f32
  }
}

// ---------------------------------------------------------------------------
// Kernel 2: dst MLP (concat(mesh, agg) -> MLP -> LN -> +mesh residual)
// ---------------------------------------------------------------------------
__global__ __launch_bounds__(128) void dst_mlp_kernel(
    const float* __restrict__ mesh, const float* __restrict__ agg,
    const bf16_t* __restrict__ W1t, const float* __restrict__ b1,
    const bf16_t* __restrict__ W2t, const float* __restrict__ b2,
    const float* __restrict__ g, const float* __restrict__ beta,
    float* __restrict__ mesh_out) {
  __shared__ alignas(32) unsigned char smem[98304];
  __shared__ float red[256];
  bf16_t* Hsh = (bf16_t*)smem;               // [32][512]
  bf16_t* Ash = (bf16_t*)(smem + 32768);     // [32][1024]
  float*  Ysh = (float*)(smem + 32768);      // [32][512] (aliases Ash, Ash dead)

  const int tid = threadIdx.x;
  const int r0 = blockIdx.x * 32;

  for (int j = tid; j < 32 * 256; j += 128) {     // 256 float4 per row
    int row = j >> 8, c4 = j & 255;
    int col = c4 * 4;
    int rr = min(r0 + row, NM_ - 1);
    const float* src = (col < 512) ? mesh + (size_t)rr * D_ + col
                                   : agg  + (size_t)rr * D_ + (col - 512);
    float4 f = *(const float4*)src;
    v4bf o = { (bf16_t)f.x, (bf16_t)f.y, (bf16_t)f.z, (bf16_t)f.w };
    *(v4bf*)(Ash + row * 1024 + col) = o;
  }
  __syncthreads();

  mlp_core<1024>(Ash, Hsh, Ysh, W1t, b1, W2t, b2, tid);

  float2 st = ln_stats32(Ysh, red, tid);
  const int row = tid >> 2, seg = tid & 3;
  const int grow = r0 + row;
  if (grow < NM_) {
#pragma unroll 4
    for (int c = 0; c < 128; ++c) {
      int col = seg * 128 + c;
      float v = (Ysh[row * D_ + col] - st.x) * st.y * g[col] + beta[col];
      mesh_out[(size_t)grow * D_ + col] = mesh[(size_t)grow * D_ + col] + v;
    }
  }
}

// ---------------------------------------------------------------------------
// Kernel 3: src MLP (grid -> MLP -> LN -> +grid residual)
// ---------------------------------------------------------------------------
__global__ __launch_bounds__(128) void src_mlp_kernel(
    const float* __restrict__ grid, const bf16_t* __restrict__ W1t,
    const float* __restrict__ b1, const bf16_t* __restrict__ W2t,
    const float* __restrict__ b2, const float* __restrict__ g,
    const float* __restrict__ beta, float* __restrict__ grid_out) {
  __shared__ alignas(32) unsigned char smem[98304];
  __shared__ float red[256];
  bf16_t* Hsh = (bf16_t*)smem;               // [32][512]
  bf16_t* Ash = (bf16_t*)(smem + 32768);     // [32][512]
  float*  Ysh = (float*)(smem + 32768);      // [32][512] (aliases Ash, Ash dead)

  const int tid = threadIdx.x;
  const int r0 = blockIdx.x * 32;

  for (int j = tid; j < 32 * 128; j += 128) {     // 128 float4 per row
    int row = j >> 7, c4 = j & 127;
    int col = c4 * 4;
    int rr = min(r0 + row, NG_ - 1);
    float4 f = *(const float4*)(grid + (size_t)rr * D_ + col);
    v4bf o = { (bf16_t)f.x, (bf16_t)f.y, (bf16_t)f.z, (bf16_t)f.w };
    *(v4bf*)(Ash + row * 512 + col) = o;
  }
  __syncthreads();

  mlp_core<512>(Ash, Hsh, Ysh, W1t, b1, W2t, b2, tid);

  float2 st = ln_stats32(Ysh, red, tid);
  const int row = tid >> 2, seg = tid & 3;
  const int grow = r0 + row;
  if (grow < NG_) {
#pragma unroll 4
    for (int c = 0; c < 128; ++c) {
      int col = seg * 128 + c;
      float v = (Ysh[row * D_ + col] - st.x) * st.y * g[col] + beta[col];
      grid_out[(size_t)grow * D_ + col] = grid[(size_t)grow * D_ + col] + v;
    }
  }
}

// ---------------------------------------------------------------------------
extern "C" void kernel_launch(void* const* d_in, const int* in_sizes, int n_in,
                              void* d_out, int out_size, void* d_ws, size_t ws_size,
                              hipStream_t stream) {
  (void)in_sizes; (void)n_in; (void)out_size; (void)ws_size;
  const float* g2m   = (const float*)d_in[0];
  const float* grid  = (const float*)d_in[1];
  const float* mesh  = (const float*)d_in[2];
  const int*   sidx  = (const int*)d_in[3];
  const int*   didx  = (const int*)d_in[4];
  const float* eW1 = (const float*)d_in[5],  *eb1 = (const float*)d_in[6];
  const float* eW2 = (const float*)d_in[7],  *eb2 = (const float*)d_in[8];
  const float* eg  = (const float*)d_in[9],  *ebt = (const float*)d_in[10];
  const float* sW1 = (const float*)d_in[11], *sb1 = (const float*)d_in[12];
  const float* sW2 = (const float*)d_in[13], *sb2 = (const float*)d_in[14];
  const float* sg  = (const float*)d_in[15], *sbt = (const float*)d_in[16];
  const float* dW1 = (const float*)d_in[17], *db1 = (const float*)d_in[18];
  const float* dW2 = (const float*)d_in[19], *db2 = (const float*)d_in[20];
  const float* dg  = (const float*)d_in[21], *dbt = (const float*)d_in[22];

  char* ws = (char*)d_ws;
  bf16_t* eW1t = (bf16_t*)(ws + OFF_EW1);
  bf16_t* eW2t = (bf16_t*)(ws + OFF_EW2);
  bf16_t* sW1t = (bf16_t*)(ws + OFF_SW1);
  bf16_t* sW2t = (bf16_t*)(ws + OFF_SW2);
  bf16_t* dW1t = (bf16_t*)(ws + OFF_DW1);
  bf16_t* dW2t = (bf16_t*)(ws + OFF_DW2);
  float*  agg  = (float*)(ws + OFF_AGG);

  float* grid_out = (float*)d_out;                       // [NG,512]
  float* mesh_out = (float*)d_out + (size_t)NG_ * D_;    // [NM,512]

  // weights -> bf16, transposed to [N][K]
  convert_wt_kernel<<<(1536 * 512 + 255) / 256, 256, 0, stream>>>(eW1, eW1t, 1536, 512);
  convert_wt_kernel<<<(512 * 512 + 255) / 256, 256, 0, stream>>>(eW2, eW2t, 512, 512);
  convert_wt_kernel<<<(512 * 512 + 255) / 256, 256, 0, stream>>>(sW1, sW1t, 512, 512);
  convert_wt_kernel<<<(512 * 512 + 255) / 256, 256, 0, stream>>>(sW2, sW2t, 512, 512);
  convert_wt_kernel<<<(1024 * 512 + 255) / 256, 256, 0, stream>>>(dW1, dW1t, 1024, 512);
  convert_wt_kernel<<<(512 * 512 + 255) / 256, 256, 0, stream>>>(dW2, dW2t, 512, 512);

  zero_kernel<<<(NM_ * D_ + 255) / 256, 256, 0, stream>>>(agg, NM_ * D_);

  edge_mlp_kernel<<<E_ / 32, 128, 0, stream>>>(g2m, grid, mesh, sidx, didx,
                                               eW1t, eb1, eW2t, eb2, eg, ebt, agg);
  dst_mlp_kernel<<<(NM_ + 31) / 32, 128, 0, stream>>>(mesh, agg, dW1t, db1, dW2t, db2,
                                                      dg, dbt, mesh_out);
  src_mlp_kernel<<<(NG_ + 31) / 32, 128, 0, stream>>>(grid, sW1t, sb1, sW2t, sb2,
                                                      sg, sbt, grid_out);
}
